// DR2FWL2Conv_58617713655959
// MI455X (gfx1250) — matrix operand order repaired and checked by reading, not compiled, not run on hardware
//
#include <hip/hip_runtime.h>
#include <cstdint>
#include <cstddef>

// ---------------- problem constants (match reference) ----------------
#define CCH   64
#define E0_N  100000
#define E1_N  500000
#define E2_N  1000000
#define T_N   1000000

// ---------------- CDNA5 WMMA types ----------------
typedef __attribute__((ext_vector_type(16))) __bf16 v16bf;
typedef __attribute__((ext_vector_type(8)))  float  v8f;

union Frag {
    v16bf v;
    uint4 q[2];
    unsigned short h[16];
};

__device__ __forceinline__ unsigned short f2bf(float f) {
    unsigned u = __float_as_uint(f);
    unsigned r = ((u >> 16) & 1u) + 0x7fffu;   // round-to-nearest-even
    return (unsigned short)((u + r) >> 16);
}

// Guaranteed hardware FP32 atomic-add at device scope (no CAS fallback).
// gfx1250 default SCOPE is WGP -> must widen to DEV for cross-workgroup scatter-sum.
__device__ __forceinline__ void global_fadd_dev(float* p, float v) {
    asm volatile("global_atomic_add_f32 %0, %1, off scope:SCOPE_DEV"
                 :: "v"((unsigned long long)(uintptr_t)p), "v"(v)
                 : "memory");
}

// ---------------- small utility kernels ----------------
__global__ void zero_f32_kernel(float* __restrict__ p, long n) {
    long i = (long)blockIdx.x * blockDim.x + threadIdx.x;
    long stride = (long)gridDim.x * blockDim.x;
    for (; i < n; i += stride) p[i] = 0.0f;
}

__global__ void cvt_bf16_kernel(const float* __restrict__ x, unsigned short* __restrict__ y, long n) {
    long i = (long)blockIdx.x * blockDim.x + threadIdx.x;
    long stride = (long)gridDim.x * blockDim.x;
    for (; i < n; i += stride) y[i] = f2bf(x[i]);
}

// W is [K][N] row-major f32 -> Wt is [N][K] bf16 (so B-fragments are contiguous in K)
__global__ void xpose_bf16_kernel(const float* __restrict__ W, unsigned short* __restrict__ Wt,
                                  int K, int N) {
    int i = blockIdx.x * blockDim.x + threadIdx.x;
    if (i < K * N) {
        int k = i / N, n = i % N;
        Wt[(size_t)n * K + k] = f2bf(W[i]);
    }
}

// ---------------- aggregation kernel ----------------
// h = relu(concat(Xa[ia], Xb[ib]) @ W + bias); atomically scatter-add h into accOut[io].
// One wave handles 16 triangle rows; GEMM is [16 x 128] @ [128 x 64] via
// 4 K-chunks (2 from Xa, 2 from Xb) x 4 N-tiles of v_wmma_f32_16x16x32_bf16.
// Weights staged in LDS once per block; B-fragments come from ds_load_b128.
__global__ __launch_bounds__(256) void agg_kernel(
    const unsigned short* __restrict__ Xa, const unsigned short* __restrict__ Xb,
    const int* __restrict__ ia, const int* __restrict__ ib, const int* __restrict__ io,
    int nTri,
    const unsigned short* __restrict__ Wt /* [64][128] bf16 */,
    const float* __restrict__ bias, float* __restrict__ accOut)
{
    __shared__ unsigned short sW[64 * 128];      // 16 KB
    for (int i = threadIdx.x; i < 64 * 128 / 8; i += blockDim.x)
        ((uint4*)sW)[i] = ((const uint4*)Wt)[i];
    __syncthreads();

    int wave = (int)((blockIdx.x * blockDim.x + threadIdx.x) >> 5);
    int base = wave * 16;
    if (base >= nTri) return;
    int l  = threadIdx.x & 31;
    int ln = l & 15;          // N-lane (also A-row index)
    int lh = l >> 4;          // lane half

    int ra = ia[base + ln];
    int rb = ib[base + ln];

    // A fragments (16x32 bf16): lane holds row=ln, K = [lh*8, +8) and [lh*8+16, +8)
    Frag A[4];
#pragma unroll
    for (int kc = 0; kc < 2; ++kc) {
        const uint4* pa = (const uint4*)(Xa + (size_t)ra * CCH + kc * 32 + lh * 8);
        A[kc].q[0]     = pa[0];
        A[kc].q[1]     = pa[2];            // +16 elements = +32B = +2 uint4
        const uint4* pb = (const uint4*)(Xb + (size_t)rb * CCH + kc * 32 + lh * 8);
        A[2 + kc].q[0] = pb[0];
        A[2 + kc].q[1] = pb[2];
    }

    v8f d[4];
#pragma unroll
    for (int nt = 0; nt < 4; ++nt)
#pragma unroll
        for (int j = 0; j < 8; ++j) d[nt][j] = 0.0f;

#pragma unroll
    for (int kc = 0; kc < 4; ++kc) {
#pragma unroll
        for (int nt = 0; nt < 4; ++nt) {
            // B fragment (32x16 bf16): lane holds col n, K = [lh*8, +8) and [lh*8+16, +8)
            Frag B;
            const uint4* p = (const uint4*)(sW + (nt * 16 + ln) * 128 + kc * 32 + lh * 8);
            B.q[0] = p[0];
            B.q[1] = p[2];
            d[nt] = __builtin_amdgcn_wmma_f32_16x16x32_bf16(
                false, A[kc].v, false, B.v, (short)0, d[nt], false, false);
        }
    }

    // epilogue: bias + relu + atomic scatter. D layout: lane l, vgpr r -> (M=r+8*lh, N=nt*16+ln)
    int oidx[8];
#pragma unroll
    for (int r = 0; r < 8; ++r) oidx[r] = io[base + r + 8 * lh];

#pragma unroll
    for (int nt = 0; nt < 4; ++nt) {
        int n = nt * 16 + ln;
        float bn = bias[n];
#pragma unroll
        for (int r = 0; r < 8; ++r) {
            float v = d[nt][r] + bn;
            v = v > 0.0f ? v : 0.0f;
            global_fadd_dev(accOut + (size_t)oidx[r] * CCH + n, v);
        }
    }
}

// ---------------- bias + LayerNorm (+optional ReLU) on D-register layout ----------------
// Rows of a 16x64 tile live within fixed 16-lane groups, so row stats reduce with shfl_xor<16.
__device__ __forceinline__ void bias_ln_inplace(
    v8f (&d)[4], const float* __restrict__ bias,
    const float* __restrict__ g, const float* __restrict__ be,
    int ln, bool doRelu)
{
    float s[8], ss[8];
#pragma unroll
    for (int r = 0; r < 8; ++r) { s[r] = 0.0f; ss[r] = 0.0f; }
#pragma unroll
    for (int nt = 0; nt < 4; ++nt) {
        float bv = bias[nt * 16 + ln];
#pragma unroll
        for (int r = 0; r < 8; ++r) {
            float v = d[nt][r] + bv;
            d[nt][r] = v;
            s[r] += v;
            ss[r] += v * v;
        }
    }
#pragma unroll
    for (int m = 1; m <= 8; m <<= 1) {
#pragma unroll
        for (int r = 0; r < 8; ++r) {
            s[r]  += __shfl_xor(s[r],  m, 32);
            ss[r] += __shfl_xor(ss[r], m, 32);
        }
    }
    float mean[8], rstd[8];
#pragma unroll
    for (int r = 0; r < 8; ++r) {
        mean[r] = s[r] * (1.0f / 64.0f);
        float var = ss[r] * (1.0f / 64.0f) - mean[r] * mean[r];
        rstd[r] = rsqrtf(var + 1e-5f);
    }
#pragma unroll
    for (int nt = 0; nt < 4; ++nt) {
        int n = nt * 16 + ln;
        float gv = g[n], bev = be[n];
#pragma unroll
        for (int r = 0; r < 8; ++r) {
            float v = (d[nt][r] - mean[r]) * rstd[r] * gv + bev;
            if (doRelu) v = v > 0.0f ? v : 0.0f;
            d[nt][r] = v;
        }
    }
}

// ---------------- fused residual + MLP + LN kernel ----------------
// x = ea + (inv ? 0.5*(acc + acc[inv]) : acc)
// out = LN( relu(LN(x@W1 + b1)) @ W2 + b2 )
// One wave per 16-row tile. W1/W2 staged in LDS per block; a per-wave LDS slice
// is used for the D->A relayout between the two GEMMs.
__global__ __launch_bounds__(256) void mlp_kernel(
    const float* __restrict__ ea, const float* __restrict__ acc,
    const int* __restrict__ inv, int nRows,
    const unsigned short* __restrict__ W1t, const float* __restrict__ b1,
    const float* __restrict__ g1, const float* __restrict__ be1,
    const unsigned short* __restrict__ W2t, const float* __restrict__ b2,
    const float* __restrict__ gn, const float* __restrict__ bn,
    float* __restrict__ out)
{
    __shared__ unsigned short sW1[64 * 64];      // 8 KB
    __shared__ unsigned short sW2[64 * 64];      // 8 KB
    __shared__ float sX[8 * 16 * CCH];           // 32 KB: 4 KB per wave
    for (int i = threadIdx.x; i < 64 * 64 / 8; i += blockDim.x) {
        ((uint4*)sW1)[i] = ((const uint4*)W1t)[i];
        ((uint4*)sW2)[i] = ((const uint4*)W2t)[i];
    }
    __syncthreads();

    int waveInBlk = threadIdx.x >> 5;
    int wave = (int)((blockIdx.x * blockDim.x + threadIdx.x) >> 5);
    int base = wave * 16;
    if (base >= nRows) return;
    int l  = threadIdx.x & 31;
    int ln = l & 15;
    int lh = l >> 4;
    float* w = sX + waveInBlk * (16 * CCH);

    // ---- build bf16 A fragments of x directly from global ----
    size_t gr = (size_t)(base + ln) * CCH;
    bool haveInv = (inv != nullptr);
    size_t gi = haveInv ? (size_t)inv[base + ln] * CCH : 0;

    Frag A[2];
#pragma unroll
    for (int kc = 0; kc < 2; ++kc) {
#pragma unroll
        for (int sub = 0; sub < 2; ++sub) {
            int off = kc * 32 + lh * 8 + sub * 16;
#pragma unroll
            for (int j = 0; j < 8; ++j) {
                float a = acc[gr + off + j];
                if (haveInv) a = 0.5f * (a + acc[gi + off + j]);
                A[kc].h[sub * 8 + j] = f2bf(ea[gr + off + j] + a);
            }
        }
    }

    // ---- GEMM1: [16x64] @ [64x64] ----
    v8f d[4];
#pragma unroll
    for (int nt = 0; nt < 4; ++nt)
#pragma unroll
        for (int j = 0; j < 8; ++j) d[nt][j] = 0.0f;
#pragma unroll
    for (int kc = 0; kc < 2; ++kc) {
#pragma unroll
        for (int nt = 0; nt < 4; ++nt) {
            Frag B;
            const uint4* p = (const uint4*)(sW1 + (nt * 16 + ln) * CCH + kc * 32 + lh * 8);
            B.q[0] = p[0];
            B.q[1] = p[2];
            d[nt] = __builtin_amdgcn_wmma_f32_16x16x32_bf16(
                false, A[kc].v, false, B.v, (short)0, d[nt], false, false);
        }
    }

    // ---- LN + ReLU, then relayout D -> A via the wave's LDS slice ----
    bias_ln_inplace(d, b1, g1, be1, ln, /*relu=*/true);

#pragma unroll
    for (int nt = 0; nt < 4; ++nt) {
        int n = nt * 16 + ln;
#pragma unroll
        for (int r = 0; r < 8; ++r)
            w[(r + 8 * lh) * CCH + n] = d[nt][r];
    }
    __builtin_amdgcn_wave_barrier();   // LDS is in-order per wave; just stop compiler reordering

    Frag A2[2];
#pragma unroll
    for (int kc = 0; kc < 2; ++kc) {
#pragma unroll
        for (int sub = 0; sub < 2; ++sub) {
            int off = kc * 32 + lh * 8 + sub * 16;
#pragma unroll
            for (int j = 0; j < 8; ++j)
                A2[kc].h[sub * 8 + j] = f2bf(w[ln * CCH + off + j]);
        }
    }

    // ---- GEMM2 ----
#pragma unroll
    for (int nt = 0; nt < 4; ++nt)
#pragma unroll
        for (int j = 0; j < 8; ++j) d[nt][j] = 0.0f;
#pragma unroll
    for (int kc = 0; kc < 2; ++kc) {
#pragma unroll
        for (int nt = 0; nt < 4; ++nt) {
            Frag B;
            const uint4* p = (const uint4*)(sW2 + (nt * 16 + ln) * CCH + kc * 32 + lh * 8);
            B.q[0] = p[0];
            B.q[1] = p[2];
            d[nt] = __builtin_amdgcn_wmma_f32_16x16x32_bf16(
                false, A2[kc].v, false, B.v, (short)0, d[nt], false, false);
        }
    }

    // ---- final LN, store ----
    bias_ln_inplace(d, b2, gn, bn, ln, /*relu=*/false);
#pragma unroll
    for (int nt = 0; nt < 4; ++nt) {
        int n = nt * 16 + ln;
#pragma unroll
        for (int r = 0; r < 8; ++r)
            out[(size_t)(base + r + 8 * lh) * CCH + n] = d[nt][r];
    }
}

// ---------------- host launcher ----------------
extern "C" void kernel_launch(void* const* d_in, const int* in_sizes, int n_in,
                              void* d_out, int out_size, void* d_ws, size_t ws_size,
                              hipStream_t stream)
{
    (void)in_sizes; (void)n_in; (void)out_size; (void)ws_size;

    const float* ea0 = (const float*)d_in[0];
    const float* ea1 = (const float*)d_in[1];
    const float* ea2 = (const float*)d_in[2];
    const int* t011 = (const int*)d_in[6];
    const int* t111 = (const int*)d_in[7];
    const int* t112 = (const int*)d_in[8];
    const int* t122 = (const int*)d_in[9];
    const int* t222 = (const int*)d_in[10];
    const int* inv1 = (const int*)d_in[11];
    const int* inv2 = (const int*)d_in[12];

    // per-distance parameter groups: [Wa, ba, W1, b1, g, be, W2, b2, norm_g, norm_b]
    const float* P[3][10];
    for (int i = 0; i < 3; ++i)
        for (int j = 0; j < 10; ++j)
            P[i][j] = (const float*)d_in[13 + i * 10 + j];

    // ---- carve workspace ----
    char* ws = (char*)d_ws;
    size_t off = 0;
    auto alloc = [&](size_t bytes) -> void* {
        off = (off + 255) & ~(size_t)255;
        void* p = ws + off;
        off += bytes;
        return p;
    };
    unsigned short* ea0b = (unsigned short*)alloc((size_t)E0_N * CCH * 2);
    unsigned short* ea1b = (unsigned short*)alloc((size_t)E1_N * CCH * 2);
    unsigned short* ea2b = (unsigned short*)alloc((size_t)E2_N * CCH * 2);
    long accTotal = (long)(E0_N + E1_N + E2_N) * CCH;
    float* accAll = (float*)alloc((size_t)accTotal * 4);
    float* acc0 = accAll;
    float* acc1 = acc0 + (size_t)E0_N * CCH;
    float* acc2 = acc1 + (size_t)E1_N * CCH;
    unsigned short* WaT[3], *W1T[3], *W2T[3];
    for (int i = 0; i < 3; ++i) {
        WaT[i] = (unsigned short*)alloc((size_t)64 * 128 * 2);
        W1T[i] = (unsigned short*)alloc((size_t)64 * 64 * 2);
        W2T[i] = (unsigned short*)alloc((size_t)64 * 64 * 2);
    }

    // ---- stage 1: zero accumulators, bf16 casts, weight transposes ----
    zero_f32_kernel<<<2048, 256, 0, stream>>>(accAll, accTotal);
    cvt_bf16_kernel<<<2048, 256, 0, stream>>>(ea0, ea0b, (long)E0_N * CCH);
    cvt_bf16_kernel<<<2048, 256, 0, stream>>>(ea1, ea1b, (long)E1_N * CCH);
    cvt_bf16_kernel<<<2048, 256, 0, stream>>>(ea2, ea2b, (long)E2_N * CCH);
    for (int i = 0; i < 3; ++i) {
        xpose_bf16_kernel<<<(128 * 64 + 255) / 256, 256, 0, stream>>>(P[i][0], WaT[i], 128, 64);
        xpose_bf16_kernel<<<(64 * 64 + 255) / 256, 256, 0, stream>>>(P[i][2], W1T[i], 64, 64);
        xpose_bf16_kernel<<<(64 * 64 + 255) / 256, 256, 0, stream>>>(P[i][6], W2T[i], 64, 64);
    }

    // triangle index rows
    const int* ij011 = t011, *ik011 = t011 + T_N, *kj011 = t011 + 2 * T_N;
    const int* ij111 = t111, *ik111 = t111 + T_N, *kj111 = t111 + 2 * T_N;
    const int* ij112 = t112, *ik112 = t112 + T_N, *kj112 = t112 + 2 * T_N;
    const int* ij122 = t122, *ik122 = t122 + T_N, *kj122 = t122 + 2 * T_N;
    const int* ij222 = t222, *ik222 = t222 + T_N, *kj222 = t222 + 2 * T_N;

    int aggBlocks = ((T_N / 16) + 7) / 8;   // 8 waves (16 rows each) per 256-thread block

    // ---- stage 2: all aggregations (symmetrization is linear -> accumulate then sym once) ----
    agg_kernel<<<aggBlocks, 256, 0, stream>>>(ea1b, ea1b, ik011, kj011, ij011, T_N, WaT[0], P[0][1], acc0);
    agg_kernel<<<aggBlocks, 256, 0, stream>>>(ea0b, ea1b, ij011, kj011, ik011, T_N, WaT[1], P[1][1], acc1);
    agg_kernel<<<aggBlocks, 256, 0, stream>>>(ea1b, ea1b, ik111, kj111, ij111, T_N, WaT[1], P[1][1], acc1);
    agg_kernel<<<aggBlocks, 256, 0, stream>>>(ea1b, ea2b, ik112, kj112, ij112, T_N, WaT[1], P[1][1], acc1);
    agg_kernel<<<aggBlocks, 256, 0, stream>>>(ea2b, ea2b, ik122, kj122, ij122, T_N, WaT[1], P[1][1], acc1);
    agg_kernel<<<aggBlocks, 256, 0, stream>>>(ea1b, ea1b, ij112, ik112, kj112, T_N, WaT[2], P[2][1], acc2);
    agg_kernel<<<aggBlocks, 256, 0, stream>>>(ea1b, ea2b, ij122, kj122, ik122, T_N, WaT[2], P[2][1], acc2);
    agg_kernel<<<aggBlocks, 256, 0, stream>>>(ea2b, ea2b, ik222, kj222, ij222, T_N, WaT[2], P[2][1], acc2);

    // ---- stage 3: residual + MLP + LN per edge set ----
    float* out0 = (float*)d_out;
    float* out1 = out0 + (size_t)E0_N * CCH;
    float* out2 = out1 + (size_t)E1_N * CCH;

    int b0 = ((E0_N / 16) + 7) / 8;
    int b1blocks = ((E1_N / 16) + 7) / 8;
    int b2blocks = ((E2_N / 16) + 7) / 8;
    mlp_kernel<<<b0, 256, 0, stream>>>(ea0, acc0, nullptr, E0_N,
        W1T[0], P[0][3], P[0][4], P[0][5], W2T[0], P[0][7], P[0][8], P[0][9], out0);
    mlp_kernel<<<b1blocks, 256, 0, stream>>>(ea1, acc1, inv1, E1_N,
        W1T[1], P[1][3], P[1][4], P[1][5], W2T[1], P[1][7], P[1][8], P[1][9], out1);
    mlp_kernel<<<b2blocks, 256, 0, stream>>>(ea2, acc2, inv2, E2_N,
        W1T[2], P[2][3], P[2][4], P[2][5], W2T[2], P[2][7], P[2][8], P[2][9], out2);
}
